// Attention_org_42125039239282
// MI455X (gfx1250) — matrix operand order repaired and checked by reading, not compiled, not run on hardware
//
#include <hip/hip_runtime.h>
#include <math.h>
#include <stdint.h>

typedef __attribute__((ext_vector_type(16))) _Float16 v16h;
typedef __attribute__((ext_vector_type(8)))  _Float16 v8h;
typedef __attribute__((ext_vector_type(8)))  float    v8f;
typedef __attribute__((ext_vector_type(4)))  float    v4f;
typedef unsigned int u32x4 __attribute__((ext_vector_type(4)));
typedef int          i32x8 __attribute__((ext_vector_type(8)));
typedef int          i32x4 __attribute__((ext_vector_type(4)));

static constexpr int LP  = 40;  // pitch (halves) for K-tile 32 path (80B rows)
static constexpr int LP2 = 72;  // pitch (halves) for K-tile 64 TDM path (128B + 16B pad = 144B)

// ---------------------------------------------------------------------------
// TDM: DMA one [64 rows x 64 halves] f16 tile (row stride lda elems) into LDS
// with 16B padding every 128B (-> 144B LDS pitch). D# per ISA 8.3/8.4.
// ---------------------------------------------------------------------------
__device__ __forceinline__ void tdm_load_tile_f16(const _Float16* g, unsigned lds_off, int lda)
{
  unsigned long long ga = (unsigned long long)(uintptr_t)g;
  u32x4 g0;
  g0[0] = 1u;                                            // count=1, user mode
  g0[1] = lds_off;                                       // lds_addr
  g0[2] = (unsigned)ga;                                  // global_addr[31:0]
  g0[3] = (unsigned)((ga >> 32) & 0x01FFFFFFu) | (2u << 30); // global_addr[56:32] | type=2
  i32x8 g1;
  // data_size=1(2B), pad_enable, pad_interval=4(32 DW=128B), pad_amount=3(4 DW=16B)
  g1[0] = (1 << 16) | (1 << 20) | (4 << 22) | (3 << 25);
  g1[1] = (int)(((unsigned)lda & 0xFFFFu) << 16);        // tensor_dim0[15:0]
  g1[2] = (int)(((unsigned)lda >> 16) | (64u << 16));    // tensor_dim0[31:16] | tensor_dim1[15:0]=64
  g1[3] = (int)(64u << 16);                              // tensor_dim1[31:16]=0 | tile_dim0=64
  g1[4] = 64;                                            // tile_dim1=64 | tile_dim2=0
  g1[5] = lda;                                           // tensor_dim0_stride[31:0]
  g1[6] = 0;                                             // stride[47:32] | dim1_stride lo
  g1[7] = 0;
  i32x4 z4 = { 0, 0, 0, 0 };
  i32x8 z8 = { 0, 0, 0, 0, 0, 0, 0, 0 };
  __builtin_amdgcn_tensor_load_to_lds(g0, g1, z4, z4, z8, 0);
}

// ---------------------------------------------------------------------------
// f16-input batched GEMM via TDM + WMMA.
// A: [M][K] f16 row-major.  B: [N][K] f16 row-major (column operand).
// C: [M][N] f32, C = alpha * A x B^T(layout-wise).  K-tile 64, tile 64x64.
// ---------------------------------------------------------------------------
__global__ __launch_bounds__(128)
void gemm_h_tdm_wmma(const _Float16* __restrict__ A, const _Float16* __restrict__ B,
                     float* __restrict__ C, int M, int N, int K,
                     long long sA, long long sB, long long sC, float alpha)
{
  __shared__ _Float16 As[64 * LP2];
  __shared__ _Float16 Bs[64 * LP2];

  A += (long long)blockIdx.z * sA;
  B += (long long)blockIdx.z * sB;
  C += (long long)blockIdx.z * sC;

  const int m0   = blockIdx.y << 6;
  const int n0   = blockIdx.x << 6;
  const int tid  = threadIdx.x;
  const int lane = tid & 31;
  const int wave = tid >> 5;
  const int wm   = (wave >> 1) << 5;
  const int wn   = (wave & 1) << 5;
  const int lmod = lane & 15;
  const int lhi  = lane >> 4;

  const unsigned ldsA = (unsigned)(uintptr_t)&As[0];   // generic LDS ptr low 32b = LDS offset
  const unsigned ldsB = (unsigned)(uintptr_t)&Bs[0];

  v8f acc[2][2] = {};

  for (int kt = 0; kt < K; kt += 64) {
    if (wave == 0)
      tdm_load_tile_f16(A + (long long)m0 * K + kt, ldsA, K);
    else if (wave == 1)
      tdm_load_tile_f16(B + (long long)n0 * K + kt, ldsB, K);
    __builtin_amdgcn_s_wait_tensorcnt(0);
    __syncthreads();

    #pragma unroll
    for (int s = 0; s < 2; ++s) {
      union HV { v16h v; v8h h[2]; };
      HV a0, a1, b0, b1;
      const _Float16* ap = As + (wm + lmod) * LP2 + (s << 5) + (lhi << 3);
      a0.h[0] = *(const v8h*)ap;          a0.h[1] = *(const v8h*)(ap + 16);
      const _Float16* ap1 = ap + (LP2 << 4);
      a1.h[0] = *(const v8h*)ap1;         a1.h[1] = *(const v8h*)(ap1 + 16);
      const _Float16* bp = Bs + (wn + lmod) * LP2 + (s << 5) + (lhi << 3);
      b0.h[0] = *(const v8h*)bp;          b0.h[1] = *(const v8h*)(bp + 16);
      const _Float16* bp1 = bp + (LP2 << 4);
      b1.h[0] = *(const v8h*)bp1;         b1.h[1] = *(const v8h*)(bp1 + 16);
      acc[0][0] = __builtin_amdgcn_wmma_f32_16x16x32_f16(false, a0.v, false, b0.v, (short)0, acc[0][0], false, false);
      acc[0][1] = __builtin_amdgcn_wmma_f32_16x16x32_f16(false, a0.v, false, b1.v, (short)0, acc[0][1], false, false);
      acc[1][0] = __builtin_amdgcn_wmma_f32_16x16x32_f16(false, a1.v, false, b0.v, (short)0, acc[1][0], false, false);
      acc[1][1] = __builtin_amdgcn_wmma_f32_16x16x32_f16(false, a1.v, false, b1.v, (short)0, acc[1][1], false, false);
    }
    __syncthreads();
  }

  #pragma unroll
  for (int i = 0; i < 2; ++i)
    #pragma unroll
    for (int j = 0; j < 2; ++j) {
      const int row = m0 + wm + (i << 4) + (lhi << 3);
      const int col = n0 + wn + (j << 4) + lmod;
      float* cp = C + (long long)row * N + col;
      #pragma unroll
      for (int r = 0; r < 8; ++r)
        cp[(long long)r * N] = alpha * acc[i][j][r];
    }
}

// ---------------------------------------------------------------------------
// fp32-input batched GEMM (weight projections): C[M,N] = A[M,K] x B[K,N].
// Convert-to-f16 through LDS, K-tile 32, tile 64x64, 4 waves.
// ---------------------------------------------------------------------------
__global__ __launch_bounds__(128)
void gemm_f32_wmma(const float* __restrict__ A, const float* __restrict__ B,
                   float* __restrict__ C, int M, int N, int K,
                   long long sA, long long sB, long long sC, float alpha)
{
  __shared__ _Float16 As[64 * LP];
  __shared__ _Float16 Bs[64 * LP];   // Bs[n][k]

  A += (long long)blockIdx.z * sA;
  B += (long long)blockIdx.z * sB;
  C += (long long)blockIdx.z * sC;

  const int m0   = blockIdx.y << 6;
  const int n0   = blockIdx.x << 6;
  const int tid  = threadIdx.x;
  const int lane = tid & 31;
  const int wave = tid >> 5;
  const int wm   = (wave >> 1) << 5;
  const int wn   = (wave & 1) << 5;
  const int lmod = lane & 15;
  const int lhi  = lane >> 4;

  v8f acc[2][2] = {};

  for (int kt = 0; kt < K; kt += 32) {
    {
      const int r  = tid >> 1;
      const int hh = (tid & 1) << 4;
      const float* src = A + (long long)(m0 + r) * K + kt + hh;
      _Float16* dst = As + r * LP + hh;
      #pragma unroll
      for (int i = 0; i < 16; i += 8) {
        v4f f0 = *(const v4f*)(src + i);
        v4f f1 = *(const v4f*)(src + i + 4);
        v8h h;
        h[0]=(_Float16)f0.x; h[1]=(_Float16)f0.y; h[2]=(_Float16)f0.z; h[3]=(_Float16)f0.w;
        h[4]=(_Float16)f1.x; h[5]=(_Float16)f1.y; h[6]=(_Float16)f1.z; h[7]=(_Float16)f1.w;
        *(v8h*)(dst + i) = h;
      }
      if (kt + 32 < K) __builtin_prefetch(src + 32, 0, 3);
    }
    {
      const int kk = tid >> 2;
      const int nc = (tid & 3) << 4;
      const float* src = B + (long long)(kt + kk) * N + n0 + nc;
      #pragma unroll
      for (int i = 0; i < 16; ++i)
        Bs[(nc + i) * LP + kk] = (_Float16)src[i];
      if (kt + 32 < K) __builtin_prefetch(src + (long long)32 * N, 0, 3);
    }
    __syncthreads();

    union HV { v16h v; v8h h[2]; };
    HV a0, a1, b0, b1;
    {
      const _Float16* ap = As + (wm + lmod) * LP + (lhi << 3);
      a0.h[0] = *(const v8h*)ap;          a0.h[1] = *(const v8h*)(ap + 16);
      const _Float16* ap1 = ap + (LP << 4);
      a1.h[0] = *(const v8h*)ap1;         a1.h[1] = *(const v8h*)(ap1 + 16);
      const _Float16* bp = Bs + (wn + lmod) * LP + (lhi << 3);
      b0.h[0] = *(const v8h*)bp;          b0.h[1] = *(const v8h*)(bp + 16);
      const _Float16* bp1 = bp + (LP << 4);
      b1.h[0] = *(const v8h*)bp1;         b1.h[1] = *(const v8h*)(bp1 + 16);
    }
    acc[0][0] = __builtin_amdgcn_wmma_f32_16x16x32_f16(false, a0.v, false, b0.v, (short)0, acc[0][0], false, false);
    acc[0][1] = __builtin_amdgcn_wmma_f32_16x16x32_f16(false, a0.v, false, b1.v, (short)0, acc[0][1], false, false);
    acc[1][0] = __builtin_amdgcn_wmma_f32_16x16x32_f16(false, a1.v, false, b0.v, (short)0, acc[1][0], false, false);
    acc[1][1] = __builtin_amdgcn_wmma_f32_16x16x32_f16(false, a1.v, false, b1.v, (short)0, acc[1][1], false, false);
    __syncthreads();
  }

  #pragma unroll
  for (int i = 0; i < 2; ++i)
    #pragma unroll
    for (int j = 0; j < 2; ++j) {
      const int row = m0 + wm + (i << 4) + (lhi << 3);
      const int col = n0 + wn + (j << 4) + lmod;
      float* cp = C + (long long)row * N + col;
      #pragma unroll
      for (int r = 0; r < 8; ++r)
        cp[(long long)r * N] = alpha * acc[i][j][r];
    }
}

// ---------------------------------------------------------------------------
// Depthwise 3x3, pad=1.  in/out: [B][C][32][32] f32, w: [C][9]
// ---------------------------------------------------------------------------
__global__ void dwconv3x3(const float* __restrict__ in, const float* __restrict__ w,
                          float* __restrict__ out, int C, int total)
{
  int idx = blockIdx.x * 256 + threadIdx.x;
  if (idx >= total) return;
  const int x = idx & 31, y = (idx >> 5) & 31;
  const int plane = idx >> 10;
  const int ch = plane % C;
  const float* ip = in + (long long)plane * 1024;
  const float* wp = w + ch * 9;
  float s = 0.f;
  #pragma unroll
  for (int ky = -1; ky <= 1; ++ky) {
    int yy = y + ky; if ((unsigned)yy >= 32u) continue;
    #pragma unroll
    for (int kx = -1; kx <= 1; ++kx) {
      int xx = x + kx; if ((unsigned)xx >= 32u) continue;
      s += ip[yy * 32 + xx] * wp[(ky + 1) * 3 + kx + 1];
    }
  }
  out[idx] = s;
}

// ---------------------------------------------------------------------------
// Depthwise 3x3 for V, writing f16 TRANSPOSED per (b,head): Vt[(b*4+h)][n][d]
// in: [B][3840][32][32] f32, w: [3840][9]
// ---------------------------------------------------------------------------
__global__ void dwconv3x3_t_h(const float* __restrict__ in, const float* __restrict__ w,
                              _Float16* __restrict__ outT, int total)
{
  int idx = blockIdx.x * 256 + threadIdx.x;
  if (idx >= total) return;
  const int x = idx & 31, y = (idx >> 5) & 31;
  const int plane = idx >> 10;          // b*3840 + ch
  const int ch = plane % 3840;
  const int b  = plane / 3840;
  const int hh = ch / 960, d = ch - hh * 960;
  const float* ip = in + (long long)plane * 1024;
  const float* wp = w + ch * 9;
  float s = 0.f;
  #pragma unroll
  for (int ky = -1; ky <= 1; ++ky) {
    int yy = y + ky; if ((unsigned)yy >= 32u) continue;
    #pragma unroll
    for (int kx = -1; kx <= 1; ++kx) {
      int xx = x + kx; if ((unsigned)xx >= 32u) continue;
      s += ip[yy * 32 + xx] * wp[(ky + 1) * 3 + kx + 1];
    }
  }
  outT[((long long)(b * 4 + hh) * 1024 + (y * 32 + x)) * 960 + d] = (_Float16)s;
}

// ---------------------------------------------------------------------------
// Grouped 3x3, 2 in-ch per group, pad=1.  w: [OC][2][3][3]
// ---------------------------------------------------------------------------
__global__ void gconv3x3_g2(const float* __restrict__ in, const float* __restrict__ w,
                            float* __restrict__ out, int OC, int total)
{
  int idx = blockIdx.x * 256 + threadIdx.x;
  if (idx >= total) return;
  const int x = idx & 31, y = (idx >> 5) & 31;
  const int plane = idx >> 10;
  const int o = plane % OC;
  const int b = plane / OC;
  const float* i0 = in + ((long long)b * OC + (o & ~1)) * 1024;
  const float* i1 = i0 + 1024;
  const float* wp = w + o * 18;
  float s = 0.f;
  #pragma unroll
  for (int ky = -1; ky <= 1; ++ky) {
    int yy = y + ky; if ((unsigned)yy >= 32u) continue;
    #pragma unroll
    for (int kx = -1; kx <= 1; ++kx) {
      int xx = x + kx; if ((unsigned)xx >= 32u) continue;
      const int wi = (ky + 1) * 3 + kx + 1;
      s += i0[yy * 32 + xx] * wp[wi] + i1[yy * 32 + xx] * wp[9 + wi];
    }
  }
  out[idx] = s;
}

// ---------------------------------------------------------------------------
// Row-wise L2 normalize, f32 in -> f16 out.  denom = max(||x||, 1e-12)
// ---------------------------------------------------------------------------
__global__ void l2norm_rows_h(const float* __restrict__ in, _Float16* __restrict__ out, int len)
{
  __shared__ float red[256];
  const float* p = in + (long long)blockIdx.x * len;
  _Float16* q = out + (long long)blockIdx.x * len;
  float s = 0.f;
  for (int i = threadIdx.x; i < len; i += 256) { float v = p[i]; s += v * v; }
  red[threadIdx.x] = s; __syncthreads();
  for (int st = 128; st > 0; st >>= 1) {
    if (threadIdx.x < st) red[threadIdx.x] += red[threadIdx.x + st];
    __syncthreads();
  }
  const float inv = 1.0f / fmaxf(sqrtf(red[0]), 1e-12f);
  for (int i = threadIdx.x; i < len; i += 256) q[i] = (_Float16)(p[i] * inv);
}

// ---------------------------------------------------------------------------
// InstanceNorm stats per (b,head): mean + 1/sqrt(var+eps) over c*KV elements
// ---------------------------------------------------------------------------
__global__ void in_stats(const float* __restrict__ attn, float* __restrict__ stats, int elems)
{
  __shared__ float rs[256], rq[256];
  const float* p = attn + (long long)blockIdx.x * elems;
  float s = 0.f, q = 0.f;
  for (int i = threadIdx.x; i < elems; i += 256) { float v = p[i]; s += v; q += v * v; }
  rs[threadIdx.x] = s; rq[threadIdx.x] = q; __syncthreads();
  for (int st = 128; st > 0; st >>= 1) {
    if (threadIdx.x < st) { rs[threadIdx.x] += rs[threadIdx.x + st]; rq[threadIdx.x] += rq[threadIdx.x + st]; }
    __syncthreads();
  }
  if (threadIdx.x == 0) {
    const float mu  = rs[0] / (float)elems;
    const float var = rq[0] / (float)elems - mu * mu;
    stats[blockIdx.x * 2 + 0] = mu;
    stats[blockIdx.x * 2 + 1] = 1.0f / sqrtf(var + 1e-5f);
  }
}

// ---------------------------------------------------------------------------
// (instancenorm -> softmax over KV=960) per row; f32 in, f16 probs out
// ---------------------------------------------------------------------------
__global__ void inorm_softmax_h(const float* __restrict__ attn, _Float16* __restrict__ probs,
                                const float* __restrict__ stats, int c)
{
  __shared__ float red[256];
  const int row = blockIdx.x;              // 0 .. 32*c-1
  const int bh  = row / c;
  const float mu = stats[bh * 2], rstd = stats[bh * 2 + 1];
  const float* p = attn + (long long)row * 960;
  _Float16* q = probs + (long long)row * 960;
  const int tid = threadIdx.x;

  float x[4]; float mx = -1e30f;
  #pragma unroll
  for (int j = 0; j < 4; ++j) {
    const int col = tid + j * 256;
    if (col < 960) { x[j] = (p[col] - mu) * rstd; mx = fmaxf(mx, x[j]); }
    else x[j] = -1e30f;
  }
  red[tid] = mx; __syncthreads();
  for (int st = 128; st > 0; st >>= 1) {
    if (tid < st) red[tid] = fmaxf(red[tid], red[tid + st]);
    __syncthreads();
  }
  mx = red[0]; __syncthreads();

  float s = 0.f;
  #pragma unroll
  for (int j = 0; j < 4; ++j) {
    if (tid + j * 256 < 960) { x[j] = __expf(x[j] - mx); s += x[j]; }
    else x[j] = 0.f;
  }
  red[tid] = s; __syncthreads();
  for (int st = 128; st > 0; st >>= 1) {
    if (tid < st) red[tid] += red[tid + st];
    __syncthreads();
  }
  const float inv = 1.0f / red[0];
  #pragma unroll
  for (int j = 0; j < 4; ++j) {
    const int col = tid + j * 256;
    if (col < 960) q[col] = (_Float16)(x[j] * inv);
  }
}

// ---------------------------------------------------------------------------
// Head mean: hm[b][i][n] = 0.25 * sum_h o[b][h][i][n]
// ---------------------------------------------------------------------------
__global__ void head_mean(const float* __restrict__ o, float* __restrict__ hm,
                          int chw, int total)
{
  int idx = blockIdx.x * 256 + threadIdx.x;
  if (idx >= total) return;
  const int b = idx / chw;
  const int r = idx - b * chw;
  const float* p = o + ((long long)b * 4) * chw + r;
  hm[idx] = 0.25f * (p[0] + p[(long long)chw] + p[(long long)2 * chw] + p[(long long)3 * chw]);
}

// ---------------------------------------------------------------------------
extern "C" void kernel_launch(void* const* d_in, const int* in_sizes, int n_in,
                              void* d_out, int out_size, void* d_ws, size_t ws_size,
                              hipStream_t stream)
{
  (void)in_sizes; (void)n_in; (void)out_size; (void)ws_size;
  constexpr int Bn = 8, HEADS = 4, KV = 960, HW = 1024, KVH = KV * HEADS;

  const float* emb[4] = { (const float*)d_in[0], (const float*)d_in[1],
                          (const float*)d_in[2], (const float*)d_in[3] };
  const float* emb_all = (const float*)d_in[4];
  const float* Wm[4] = { (const float*)d_in[5],  (const float*)d_in[8],
                         (const float*)d_in[11], (const float*)d_in[14] };
  const float* Wq[4] = { (const float*)d_in[6],  (const float*)d_in[9],
                         (const float*)d_in[12], (const float*)d_in[15] };
  const float* Wp[4] = { (const float*)d_in[7],  (const float*)d_in[10],
                         (const float*)d_in[13], (const float*)d_in[16] };
  const float* Wmk = (const float*)d_in[17];
  const float* Wmv = (const float*)d_in[18];
  const float* Wk  = (const float*)d_in[19];
  const float* Wv  = (const float*)d_in[20];
  float* out = (float*)d_out;

  // workspace carving (byte offsets, 256B aligned)
  char* wsp = (char*)d_ws;
  const long long S1 = (long long)Bn * KVH * HW;          // 31,457,280 elems
  float*    X   = (float*)wsp;                wsp += S1 * 4;                       // staging f32
  float*    Kb  = (float*)wsp;                wsp += S1 * 4;                       // K pre-norm f32
  _Float16* Kh  = (_Float16*)wsp;             wsp += S1 * 2;                       // normalized K f16 [b,h][960][1024]
  _Float16* Vt  = (_Float16*)wsp;             wsp += S1 * 2;                       // V f16 transposed [b,h][1024][960]
  float*    Qb  = (float*)wsp;                wsp += (long long)Bn * 2048 * HW * 4; // q post-conv f32
  _Float16* Qh  = (_Float16*)wsp;             wsp += (long long)Bn * 2048 * HW * 2; // normalized Q f16
  float*    ATT = (float*)wsp;                wsp += (long long)Bn * HEADS * 512 * KV * 4; // attn f32
  _Float16* Ph  = (_Float16*)wsp;             wsp += (long long)Bn * HEADS * 512 * KV * 2; // probs f16
  float*    Ob  = (float*)wsp;                wsp += (long long)Bn * HEADS * 512 * HW * 4; // per-head out f32
  float*    ST  = (float*)wsp;                                                    // stats (64 floats)

  const float kscale = 1.0f / sqrtf((float)KV);

  // ---- shared K = l2norm(dw3x3(Wmk @ emb_all)) -> f16 ; V = dw3x3(Wmv @ emb_all) -> f16^T ----
  gemm_f32_wmma<<<dim3(HW / 64, KVH / 64, Bn), 128, 0, stream>>>(
      Wmk, emb_all, X, KVH, HW, KV, 0, (long long)KV * HW, (long long)KVH * HW, 1.0f);
  dwconv3x3<<<(Bn * KVH * HW) / 256, 256, 0, stream>>>(X, Wk, Kb, KVH, Bn * KVH * HW);
  l2norm_rows_h<<<Bn * KVH, 256, 0, stream>>>(Kb, Kh, HW);

  gemm_f32_wmma<<<dim3(HW / 64, KVH / 64, Bn), 128, 0, stream>>>(
      Wmv, emb_all, X, KVH, HW, KV, 0, (long long)KV * HW, (long long)KVH * HW, 1.0f);
  dwconv3x3_t_h<<<(Bn * KVH * HW) / 256, 256, 0, stream>>>(X, Wv, Vt, Bn * KVH * HW);

  // ---- per-branch pipeline ----
  const int cs[4] = { 64, 128, 256, 512 };
  long long out_off = 0;
  for (int br = 0; br < 4; ++br) {
    const int c  = cs[br];
    const int c4 = c * HEADS;

    // q_pre = Wm @ emb (X), grouped 3x3 -> Qb, l2norm -> Qh
    gemm_f32_wmma<<<dim3(HW / 64, c4 / 64, Bn), 128, 0, stream>>>(
        Wm[br], emb[br], X, c4, HW, c, 0, (long long)c * HW, (long long)c4 * HW, 1.0f);
    gconv3x3_g2<<<(Bn * c4 * HW) / 256, 256, 0, stream>>>(X, Wq[br], Qb, c4, Bn * c4 * HW);
    l2norm_rows_h<<<Bn * c4, 256, 0, stream>>>(Qb, Qh, HW);

    // attn = scale * q @ k^T   (A=[c][1024] f16, B=[960][1024] f16, batched b*heads)
    gemm_h_tdm_wmma<<<dim3(KV / 64, c / 64, Bn * HEADS), 128, 0, stream>>>(
        Qh, Kh, ATT, c, KV, HW,
        (long long)c * HW, (long long)KV * HW, (long long)c * KV, kscale);

    // InstanceNorm over (c,KV) per (b,h), softmax over KV -> f16 probs
    in_stats<<<Bn * HEADS, 256, 0, stream>>>(ATT, ST, c * KV);
    inorm_softmax_h<<<Bn * HEADS * c, 256, 0, stream>>>(ATT, Ph, ST, c);

    // o = probs @ v   (A=[c][960] f16, B=Vt [1024][960] f16, batched b*heads)
    gemm_h_tdm_wmma<<<dim3(HW / 64, c / 64, Bn * HEADS), 128, 0, stream>>>(
        Ph, Vt, Ob, c, HW, KV,
        (long long)c * KV, (long long)KV * HW, (long long)c * HW, 1.0f);

    // mean over heads -> X, then out = Wp @ X
    head_mean<<<(Bn * c * HW) / 256, 256, 0, stream>>>(Ob, X, c * HW, Bn * c * HW);
    gemm_f32_wmma<<<dim3(HW / 64, c / 64, Bn), 128, 0, stream>>>(
        Wp[br], X, out + out_off, c, HW, c,
        0, (long long)c * HW, (long long)c * HW, 1.0f);

    out_off += (long long)Bn * c * HW;
  }
}
